// EGNNMultiHeadJump_42339787604764
// MI455X (gfx1250) — matrix operand order repaired and checked by reading, not compiled, not run on hardware
//
#include <hip/hip_runtime.h>
#include <hip/hip_bf16.h>
#include <math.h>

// ---------------------------------------------------------------------------
// Reference shapes: B=64, N=64, H=256, TD=256, NL=4, NAB=2, NH=8
// ---------------------------------------------------------------------------
#define OUTROW 1793   // 576 D_xt + 1 rate + 64 near + 576 auto_mean + 576 auto_std

typedef __attribute__((ext_vector_type(16))) __bf16   v16bf;
typedef __attribute__((ext_vector_type(8)))  float    v8f;
typedef __attribute__((ext_vector_type(8)))  unsigned v8u;

// ----------------------------- small helpers ------------------------------
__device__ __forceinline__ unsigned bfbits(float f) {       // f32 -> bf16 (RNE)
  unsigned u = __builtin_bit_cast(unsigned, f);
  return (u + 0x7FFFu + ((u >> 16) & 1u)) >> 16;
}
__device__ __forceinline__ unsigned packbf(float a, float b) {
  return bfbits(a) | (bfbits(b) << 16);
}
__device__ __forceinline__ __bf16 f2bf(float f) {
  unsigned short s = (unsigned short)bfbits(f);
  return __builtin_bit_cast(__bf16, s);
}
__device__ __forceinline__ float bf2f(__bf16 h) {
  unsigned short s = __builtin_bit_cast(unsigned short, h);
  unsigned u = ((unsigned)s) << 16;
  return __builtin_bit_cast(float, u);
}
// fast sigmoid / silu: native v_exp_f32 + v_rcp_f32 (bf16 matmuls bound accuracy)
__device__ __forceinline__ float sigm(float x) {
  return __builtin_amdgcn_rcpf(1.f + __expf(-x));
}
__device__ __forceinline__ float silu(float x) { return x * sigm(x); }
__device__ __forceinline__ v8f v8zero() {
  v8f z;
#pragma unroll
  for (int i = 0; i < 8; ++i) z[i] = 0.f;
  return z;
}
__device__ __forceinline__ v8f wmma_bf16(v16bf a, v16bf b, v8f c) {
  return __builtin_amdgcn_wmma_f32_16x16x32_bf16(false, a, false, b, (short)0, c, false, false);
}

// Async global->LDS byte copy (CDNA5, ASYNCcnt). 16B per active lane.
__device__ __forceinline__ void async_ld_b128(void* lds, const void* gaddr) {
  unsigned l = (unsigned)(size_t)lds;
  unsigned long long g = (unsigned long long)(size_t)gaddr;
  asm volatile("global_load_async_to_lds_b128 %0, %1, off" :: "v"(l), "v"(g) : "memory");
}
__device__ __forceinline__ void wait_async0() {
  asm volatile("s_wait_asynccnt 0" ::: "memory");
}
__device__ __forceinline__ void wait_async1() {
  asm volatile("s_wait_asynccnt 1" ::: "memory");
}

// ----- packed-pair fragment loads (dword = {bf16 K=2p, bf16 K=2p+1}) -------
__device__ __forceinline__ v16bf load_a_pk(const unsigned* base, int stride) {
  int lane = threadIdx.x & 31;
  int l = lane & 15, hi = lane >> 4;
  const unsigned* row = base + l * stride;
  v8u u;
#pragma unroll
  for (int g = 0; g < 8; ++g) {
    int pa = g + (hi ? 8 : 0);
    u[g] = row[pa];
  }
  return __builtin_bit_cast(v16bf, u);
}
__device__ __forceinline__ v16bf load_b_pk(const unsigned* base, int stride) {
  int lane = threadIdx.x & 31;
  int l = lane & 15, hi = lane >> 4;
  v8u u;
#pragma unroll
  for (int g = 0; g < 8; ++g) u[g] = base[((hi ? 8 : 0) + g) * stride + l];
  return __builtin_bit_cast(v16bf, u);
}

// ----- unpacked b16 fragment loads (general/bounds-checked path only) ------
__device__ __forceinline__ v16bf load_a_frag(const __bf16* base, int stride) {
  int lane = threadIdx.x & 31;
  int l = lane & 15, hi = lane >> 4;
  const __bf16* row = base + l * stride;
  v16bf a;
#pragma unroll
  for (int g = 0; g < 8; ++g) {
    int kb = 2 * g + (hi ? 16 : 0);
    a[2 * g] = row[kb]; a[2 * g + 1] = row[kb + 1];
  }
  return a;
}
__device__ __forceinline__ v16bf load_b_frag(const __bf16* base, int stride) {
  int lane = threadIdx.x & 31;
  int l = lane & 15, hi = lane >> 4;
  v16bf b;
#pragma unroll
  for (int g = 0; g < 8; ++g) {
    int k = (hi ? 16 : 0) + 2 * g;
    b[2 * g] = base[k * stride + l]; b[2 * g + 1] = base[(k + 1) * stride + l];
  }
  return b;
}

// ---------------------------------------------------------------------------
// Weight pre-pack: f32 W[K,N] -> dword Wp[K/2, N] = {bf16 W[2p,n], bf16 W[2p+1,n]}
// ---------------------------------------------------------------------------
__global__ void k_pack(const float* W, unsigned* Wp, int K, int N)
{
  int i = blockIdx.x * blockDim.x + threadIdx.x;
  int total = (K >> 1) * N;
  if (i >= total) return;
  int p = i / N, n = i - p * N;
  Wp[i] = packbf(W[(size_t)(2 * p) * N + n], W[(size_t)(2 * p + 1) * N + n]);
}

// ---------------------------------------------------------------------------
// FAST WMMA GEMM (M%64==0, K%32==0, N%64==0), templated branchless epilogue:
//   Y = mask? * ( res? + act?( X @ W + bias? ) );  W pre-packed bf16 pairs.
// B tiles double-buffered via global_load_async_to_lds_b128 (ASYNCcnt).
// ---------------------------------------------------------------------------
template<bool BIAS, int ACT, bool RES, bool MASK>
__global__ __launch_bounds__(256) void k_gemm_fast(
    const float* X, const unsigned* Wp, const float* bias,
    const float* residual, const float* rowmask, float* Y,
    int M, int K, int N)
{
  __shared__ alignas(16) unsigned Asp[64][20];
  __shared__ alignas(16) unsigned Bsp[2][16][72];
  int tid = threadIdx.x;
  int wid = tid >> 5, mt = wid & 3, nh = wid >> 2;
  int Mb = blockIdx.x * 64, Nb = blockIdx.y * 64;

  int ar = tid >> 2, apg = (tid & 3) * 4;                 // A: row, pair-group
  const float* xrow = X + (size_t)(Mb + ar) * K + apg * 2;
  int bp = tid >> 4, bch = (tid & 15) * 4;                // B: pair-row, col chunk
  const unsigned* wsrc = Wp + (size_t)bp * N + Nb + bch;

  async_ld_b128(&Bsp[0][bp][bch], wsrc);                  // prefetch first B tile
  v8f a0 = v8zero(), a1 = v8zero();
  for (int k0 = 0; k0 < K; k0 += 32) {
    int cur = (k0 >> 5) & 1;
    bool more = (k0 + 32) < K;
    if (more)                                              // prefetch next B tile
      async_ld_b128(&Bsp[cur ^ 1][bp][bch], wsrc + (size_t)((k0 >> 1) + 16) * N);
    const float4* x4 = (const float4*)(xrow + k0);
    float4 q0 = x4[0], q1 = x4[1];
    uint4 pk;
    pk.x = packbf(q0.x, q0.y); pk.y = packbf(q0.z, q0.w);
    pk.z = packbf(q1.x, q1.y); pk.w = packbf(q1.z, q1.w);
    *(uint4*)&Asp[ar][apg] = pk;
    if (more) wait_async1(); else wait_async0();
    __syncthreads();
    v16bf af = load_a_pk(&Asp[mt * 16][0], 20);
    v16bf b0 = load_b_pk(&Bsp[cur][0][nh * 32], 72);
    v16bf b1 = load_b_pk(&Bsp[cur][0][nh * 32 + 16], 72);
    a0 = wmma_bf16(af, b0, a0);
    a1 = wmma_bf16(af, b1, a1);
    __syncthreads();
  }

  int lane = tid & 31, l = lane & 15, hi = lane >> 4;
#pragma unroll
  for (int t = 0; t < 2; ++t) {
    v8f acc; if (t == 0) acc = a0; else acc = a1;
    int col = Nb + nh * 32 + t * 16 + l;
#pragma unroll
    for (int v = 0; v < 8; ++v) {
      int row = Mb + mt * 16 + hi * 8 + v;
      float val = acc[v];
      if (BIAS) val += bias[col];
      if (ACT == 1) val = silu(val);
      if (RES) val += residual[(size_t)row * N + col];
      if (MASK) val *= rowmask[row];
      Y[(size_t)row * N + col] = val;
    }
  }
}

// ---------------------------------------------------------------------------
// GENERAL WMMA GEMM (bounds-checked; odd K/N shapes: 13, 262, 265, 7, 13)
// ---------------------------------------------------------------------------
__global__ __launch_bounds__(256) void k_gemm(
    const float* X, const float* W, const float* bias,
    const float* residual, const float* rowmask, float* Y,
    int M, int K, int N, int act)
{
  __shared__ __bf16 As[64][40];
  __shared__ __bf16 Bs[32][72];
  int tid = threadIdx.x;
  int wid = tid >> 5, mt = wid & 3, nh = wid >> 2;
  int Mb = blockIdx.x * 64, Nb = blockIdx.y * 64;
  v8f a0 = v8zero(), a1 = v8zero();

  for (int k0 = 0; k0 < K; k0 += 32) {
    for (int i = tid; i < 64 * 32; i += 256) {
      int r = i >> 5, c = i & 31;
      int gr = Mb + r, gk = k0 + c;
      As[r][c] = (gr < M && gk < K) ? f2bf(X[(size_t)gr * K + gk]) : f2bf(0.f);
    }
    for (int i = tid; i < 32 * 64; i += 256) {
      int r = i >> 6, c = i & 63;
      int gk = k0 + r, gc = Nb + c;
      Bs[r][c] = (gk < K && gc < N) ? f2bf(W[(size_t)gk * N + gc]) : f2bf(0.f);
    }
    __syncthreads();
    v16bf af = load_a_frag(&As[mt * 16][0], 40);
    v16bf b0 = load_b_frag(&Bs[0][nh * 32], 72);
    v16bf b1 = load_b_frag(&Bs[0][nh * 32 + 16], 72);
    a0 = wmma_bf16(af, b0, a0);
    a1 = wmma_bf16(af, b1, a1);
    __syncthreads();
  }

  int lane = tid & 31, l = lane & 15, hi = lane >> 4;
#pragma unroll
  for (int t = 0; t < 2; ++t) {
    v8f acc; if (t == 0) acc = a0; else acc = a1;
    int col = Nb + nh * 32 + t * 16 + l;
    if (col >= N) continue;
#pragma unroll
    for (int v = 0; v < 8; ++v) {
      int row = Mb + mt * 16 + hi * 8 + v;
      if (row >= M) continue;
      float val = acc[v];
      if (bias) val += bias[col];
      if (act == 1) val = silu(val);
      if (residual) val += residual[(size_t)row * N + col];
      if (rowmask) val *= rowmask[row];
      Y[(size_t)row * N + col] = val;
    }
  }
}

// ---------------------------------------------------------------------------
// Fused EGNN edge MLP. One workgroup per (b,i); W2 pre-packed, async-staged.
// ---------------------------------------------------------------------------
__global__ __launch_bounds__(256) void k_edge(
    const float* Pi, const float* Pj, const float* wd, const float* wd0,
    const float* b1, const unsigned* W2p, const float* b2,
    const float* vw, const float* vb,
    const float* dmat, const float* d0mat, const float* emask,
    const float* cdiff, float* outp, int mode)
{
  __shared__ alignas(16) unsigned Tsp[64][132];   // A (packed pairs), reused for M
  __shared__ alignas(16) unsigned Wsp[16][260];   // W2 K-slice (packed pairs)
  __shared__ float hi_s[256];
  __shared__ float pr[256];
  __shared__ float red[64];

  int tid = threadIdx.x;
  int bi = blockIdx.x;           // b*64 + i
  int b = bi >> 6;

  hi_s[tid] = Pi[(size_t)bi * 256 + tid];
  __syncthreads();

  for (int idx = tid; idx < 64 * 128; idx += 256) {     // build silu(edge1) packed
    int j = idx >> 7, cp = idx & 127, c = cp * 2;
    float dv  = dmat[(size_t)bi * 64 + j];
    float d0v = d0mat[(size_t)bi * 64 + j];
    const float2 pj = *(const float2*)&Pj[(size_t)(b * 64 + j) * 256 + c];
    float x0 = hi_s[c] + pj.x + dv * wd[c] + d0v * wd0[c] + b1[c];
    float x1 = hi_s[c + 1] + pj.y + dv * wd[c + 1] + d0v * wd0[c + 1] + b1[c + 1];
    Tsp[j][cp] = packbf(silu(x0), silu(x1));
  }

  int wid = tid >> 5, lane = tid & 31, l = lane & 15, hi = lane >> 4;
  int mt = wid & 3, nb = (wid >> 2) * 128;              // 16 rows x 128 cols per wave
  v8f acc[8];
#pragma unroll
  for (int t = 0; t < 8; ++t) acc[t] = v8zero();

  for (int k0 = 0; k0 < 256; k0 += 32) {
#pragma unroll
    for (int q = 0; q < 4; ++q) {                       // async stage 16x256 dwords
      int idx = tid + q * 256;
      int p = idx >> 6, ch = (idx & 63) * 4;
      async_ld_b128(&Wsp[p][ch], W2p + (size_t)((k0 >> 1) + p) * 256 + ch);
    }
    wait_async0();
    __syncthreads();
    v16bf af = load_a_pk(&Tsp[mt * 16][k0 >> 1], 132);
#pragma unroll
    for (int t = 0; t < 8; ++t) {
      v16bf bf = load_b_pk(&Wsp[0][nb + t * 16], 260);
      acc[t] = wmma_bf16(af, bf, acc[t]);
    }
    __syncthreads();
  }
  __syncthreads();

  // write silu(acc + b2) back into Tsp (viewed as bf16[64][264])
  __bf16* Tb = (__bf16*)&Tsp[0][0];
#pragma unroll
  for (int t = 0; t < 8; ++t) {
    int c = nb + t * 16 + l;
#pragma unroll
    for (int v = 0; v < 8; ++v) {
      int j = mt * 16 + hi * 8 + v;
      Tb[j * 264 + c] = f2bf(silu(acc[t][v] + b2[c]));
    }
  }
  __syncthreads();

  {
    int j = tid >> 2, t4 = tid & 3;
    float s = 0.f;
#pragma clang loop unroll_count(8)
    for (int c = t4 * 64; c < t4 * 64 + 64; ++c) s += bf2f(Tb[j * 264 + c]) * vw[c];
    pr[tid] = s;
  }
  __syncthreads();
  if (tid < 64) {
    float s = pr[tid * 4] + pr[tid * 4 + 1] + pr[tid * 4 + 2] + pr[tid * 4 + 3];
    float em = emask[(size_t)bi * 64 + tid];
    red[tid] = (mode == 0) ? sigm(s + vb[0]) * em : s * em;
  }
  __syncthreads();

  if (mode == 0) {
    float s = 0.f;
#pragma clang loop unroll_count(8)
    for (int j = 0; j < 64; ++j) s += bf2f(Tb[j * 264 + tid]) * red[j];
    outp[(size_t)bi * 256 + tid] = s;
  } else if (tid < 3) {
    float s = 0.f;
#pragma clang loop unroll_count(8)
    for (int j = 0; j < 64; ++j) s += cdiff[((size_t)bi * 64 + j) * 3 + tid] * red[j];
    outp[(size_t)bi * 3 + tid] = s;
  }
}

// ---------------------------------------------------------------------------
// GroupNorm over (B,N,256); 32 groups of 8 channels; stats over (8ch x 64n).
// ---------------------------------------------------------------------------
__global__ __launch_bounds__(64) void k_gnorm(const float* X, const float* gw,
                                              const float* gb, float* Y, int dosilu)
{
  __shared__ float s1[64], s2[64];
  int bg = blockIdx.x; int b = bg >> 5, g = bg & 31;
  int n = threadIdx.x;
  int c0 = g * 8;
  float v[8]; float s = 0.f, q = 0.f;
#pragma unroll
  for (int c = 0; c < 8; ++c) {
    float x = X[((size_t)(b * 64 + n)) * 256 + c0 + c];
    v[c] = x; s += x; q += x * x;
  }
  s1[n] = s; s2[n] = q; __syncthreads();
  for (int st = 32; st > 0; st >>= 1) {
    if (n < st) { s1[n] += s1[n + st]; s2[n] += s2[n + st]; }
    __syncthreads();
  }
  float mu = s1[0] * (1.f / 512.f);
  float var = s2[0] * (1.f / 512.f) - mu * mu;
  float rstd = rsqrtf(var + 1e-6f);
#pragma unroll
  for (int c = 0; c < 8; ++c) {
    float y = (v[c] - mu) * rstd * gw[c0 + c] + gb[c0 + c];
    if (dosilu) y = silu(y);
    Y[((size_t)(b * 64 + n)) * 256 + c0 + c] = y;
  }
}

// ---------------------------------------------------------------------------
// Attention: one workgroup per (b, head); N=64, head_dim=32. Packed-pair WMMA.
// ---------------------------------------------------------------------------
__global__ __launch_bounds__(128) void k_attn(const float* Q, const float* Kt,
                                              const float* V, float* O)
{
  __shared__ alignas(16) unsigned qsp[64][20];   // A: rows i, pairs over d
  __shared__ alignas(16) unsigned ktp[16][72];   // B: pair-rows over d, cols j
  __shared__ alignas(16) unsigned vsp[32][36];   // B: pair-rows over j, cols d
  __shared__ float Ss[64][68];
  __shared__ alignas(16) unsigned Psp[64][36];   // A: rows i, pairs over j
  int tid = threadIdx.x;
  int bh = blockIdx.x; int b = bh >> 3, h = bh & 7; int c0 = h * 32;

  for (int idx = tid; idx < 64 * 16; idx += 128) {
    int n = idx >> 4, dp = idx & 15;
    size_t off = ((size_t)(b * 64 + n)) * 256 + c0 + 2 * dp;
    float2 q2 = *(const float2*)&Q[off];
    float2 k2 = *(const float2*)&Kt[off];
    qsp[n][dp] = packbf(q2.x, q2.y);
    ktp[dp][n] = packbf(k2.x, k2.y);
  }
  for (int idx = tid; idx < 32 * 32; idx += 128) {
    int jp = idx >> 5, d = idx & 31;
    size_t o0 = ((size_t)(b * 64 + 2 * jp)) * 256 + c0 + d;
    vsp[jp][d] = packbf(V[o0], V[o0 + 256]);
  }
  __syncthreads();

  int wid = tid >> 5, lane = tid & 31, l = lane & 15, hi = lane >> 4;
  {
    v16bf af = load_a_pk(&qsp[wid * 16][0], 20);
    const float sc = 0.17677669529663687f;  // 1/sqrt(32)
#pragma unroll
    for (int t = 0; t < 4; ++t) {
      v16bf bf = load_b_pk(&ktp[0][t * 16], 72);
      v8f acc = v8zero();
      acc = wmma_bf16(af, bf, acc);
#pragma unroll
      for (int v = 0; v < 8; ++v) Ss[wid * 16 + hi * 8 + v][t * 16 + l] = acc[v] * sc;
    }
  }
  __syncthreads();
  if (tid < 64) {
    float mx = -1e30f;
#pragma clang loop unroll_count(8)
    for (int j = 0; j < 64; ++j) mx = fmaxf(mx, Ss[tid][j]);
    float sum = 0.f;
#pragma clang loop unroll_count(8)
    for (int j = 0; j < 64; ++j) { float e = __expf(Ss[tid][j] - mx); Ss[tid][j] = e; sum += e; }
    float inv = __builtin_amdgcn_rcpf(sum);
#pragma clang loop unroll_count(8)
    for (int jp = 0; jp < 32; ++jp)
      Psp[tid][jp] = packbf(Ss[tid][2 * jp] * inv, Ss[tid][2 * jp + 1] * inv);
  }
  __syncthreads();
  {
    v8f a0 = v8zero(), a1 = v8zero();
#pragma unroll
    for (int k0 = 0; k0 < 64; k0 += 32) {
      v16bf af = load_a_pk(&Psp[wid * 16][k0 >> 1], 36);
      v16bf b0 = load_b_pk(&vsp[k0 >> 1][0], 36);
      v16bf b1 = load_b_pk(&vsp[k0 >> 1][16], 36);
      a0 = wmma_bf16(af, b0, a0);
      a1 = wmma_bf16(af, b1, a1);
    }
#pragma unroll
    for (int v = 0; v < 8; ++v) {
      int row = wid * 16 + hi * 8 + v;
      size_t off = ((size_t)(b * 64 + row)) * 256 + c0;
      O[off + l] = a0[v];
      O[off + 16 + l] = a1[v];
    }
  }
}

// ------------------------- small elementwise kernels -----------------------
__global__ void k_prep(const float* x, const float* oh, const float* ch,
                       const float* ctx, const float* ts, const int* dims,
                       float* nm, float* hin, float* xm)
{
  int b = blockIdx.x, n = threadIdx.x;
  float m = (n < dims[b]) ? 1.f : 0.f;
  nm[b * 64 + n] = m;
  size_t r = (size_t)(b * 64 + n);
  for (int k = 0; k < 3; ++k) xm[r * 3 + k] = x[r * 3 + k] * m;
  float* hv = hin + r * 13;
  for (int k = 0; k < 5; ++k) hv[k] = oh[r * 5 + k] * m;
  hv[5] = ch[r] * m;
  hv[6] = ts[b];
  for (int k = 0; k < 6; ++k) hv[7 + k] = ctx[b * 6 + k] * m;
}

__global__ void k_pair(const float* xc, const float* nm, float* dout,
                       float* cdiff, float* emask, int mode)
{
  int bi = blockIdx.x; int b = bi >> 6, i = bi & 63; int j = threadIdx.x;
  const float* xi = xc + (size_t)bi * 3;
  const float* xj = xc + (size_t)(b * 64 + j) * 3;
  float d0 = xi[0] - xj[0], d1 = xi[1] - xj[1], d2 = xi[2] - xj[2];
  float r = d0 * d0 + d1 * d1 + d2 * d2;
  dout[(size_t)bi * 64 + j] = r;
  if (mode == 0) {
    emask[(size_t)bi * 64 + j] = nm[b * 64 + i] * nm[b * 64 + j] * ((i == j) ? 0.f : 1.f);
  } else {
    float inv = __builtin_amdgcn_rcpf(sqrtf(r + 1e-8f) + 1.f);
    size_t o = ((size_t)bi * 64 + j) * 3;
    cdiff[o] = d0 * inv; cdiff[o + 1] = d1 * inv; cdiff[o + 2] = d2 * inv;
  }
}

__global__ void k_copy(const float* a, float* y, int n)
{ int i = blockIdx.x * blockDim.x + threadIdx.x; if (i < n) y[i] = a[i]; }

__global__ void k_silu(const float* a, float* y, int n)
{ int i = blockIdx.x * blockDim.x + threadIdx.x; if (i < n) y[i] = silu(a[i]); }

__global__ void k_cat2(const float* A, const float* Bp, float* Y, int rows)
{
  int i = blockIdx.x * blockDim.x + threadIdx.x;
  if (i >= rows * 512) return;
  int r = i >> 9, c = i & 511;
  Y[i] = (c < 256) ? A[(size_t)r * 256 + c] : Bp[(size_t)r * 256 + (c - 256)];
}

__global__ void k_maskmul(const float* X, const float* nm, float* Y, int n)
{ int i = blockIdx.x * blockDim.x + threadIdx.x; if (i < n) Y[i] = X[i] * nm[i >> 8]; }

__global__ void k_xupd(float* xc, const float* xagg, const float* nm, int n)
{
  int i = blockIdx.x * blockDim.x + threadIdx.x;
  if (i >= n) return;
  xc[i] = (xc[i] + xagg[i]) * nm[i / 3];
}

__global__ void k_addrow(float* R, const float* tv, int n)
{
  int i = blockIdx.x * blockDim.x + threadIdx.x;
  if (i >= n) return;
  R[i] += tv[(size_t)(i >> 14) * 256 + (i & 255)];
}

__global__ void k_tembed(const float* ts, float* emb)
{
  int b = blockIdx.x, i = threadIdx.x;  // 128
  float f = __expf(-9.210340371976184f / 127.f * (float)i);
  float a = ts[b] * 1000.f * f;
  emb[(size_t)b * 256 + i] = sinf(a);
  emb[(size_t)b * 256 + 128 + i] = cosf(a);
}

__global__ void k_mean(const float* X, float* Y)
{
  int b = blockIdx.x, c = threadIdx.x;  // 256
  float s = 0.f;
#pragma clang loop unroll_count(8)
  for (int n = 0; n < 64; ++n) s += X[((size_t)(b * 64 + n)) * 256 + c];
  Y[(size_t)b * 256 + c] = s * (1.f / 64.f);
}

__global__ __launch_bounds__(64) void k_rate(const float* m, const float* w,
                                             const float* bias, const float* fr, float* out)
{
  __shared__ float sh[64];
  int b = blockIdx.x, t = threadIdx.x;
  float s = 0.f;
  for (int c = t; c < 256; c += 64) s += m[(size_t)b * 256 + c] * w[c];
  sh[t] = s; __syncthreads();
  for (int st = 32; st > 0; st >>= 1) { if (t < st) sh[t] += sh[t + st]; __syncthreads(); }
  if (t == 0) {
    float x = sh[0] + bias[0];
    float sp = (x > 20.f) ? x : log1pf(__expf(x));
    out[(size_t)b * OUTROW + 576] = sp * fr[b];
  }
}

__global__ void k_near(const float* h, const float* w, const float* bias, float* out)
{
  int b = blockIdx.x, n = threadIdx.x;
  float s = bias[0];
#pragma clang loop unroll_count(8)
  for (int c = 0; c < 256; ++c) s += h[((size_t)(b * 64 + n)) * 256 + c] * w[c];
  out[(size_t)b * OUTROW + 577 + n] = s;
}

__global__ void k_dotrow(const float* h, const float* w, const float* bias, float* y)
{
  int b = blockIdx.x, n = threadIdx.x;
  size_t r = (size_t)(b * 64 + n);
  float s = bias[0];
#pragma clang loop unroll_count(8)
  for (int c = 0; c < 256; ++c) s += h[r * 256 + c] * w[c];
  y[r] = s;
}

__global__ __launch_bounds__(64) void k_vel(const float* xfin, const float* xm,
                                            const float* nm, const int* dims,
                                            const float* hout, float* out)
{
  __shared__ float sh[3][64];
  int b = blockIdx.x, n = threadIdx.x;
  float m = nm[b * 64 + n];
  size_t r = (size_t)(b * 64 + n);
  float v[3];
  for (int k = 0; k < 3; ++k) { v[k] = (xfin[r * 3 + k] - xm[r * 3 + k]) * m; sh[k][n] = v[k]; }
  __syncthreads();
  for (int st = 32; st > 0; st >>= 1) {
    if (n < st) for (int k = 0; k < 3; ++k) sh[k][n] += sh[k][n + st];
    __syncthreads();
  }
  float cnt = (float)dims[b];
  for (int k = 0; k < 3; ++k)
    out[(size_t)b * OUTROW + n * 3 + k] = (v[k] - sh[k][0] / cnt) * m;
  const float* ho = hout + r * 7;
  for (int q = 0; q < 5; ++q) out[(size_t)b * OUTROW + 192 + n * 5 + q] = ho[q];
  out[(size_t)b * OUTROW + 512 + n] = ho[5];
}

__global__ void k_t1cat(const float* hl, const float* hin, float* Y)
{
  int b = blockIdx.x, n = threadIdx.x;
  size_t r = (size_t)(b * 64 + n);
  float* y = Y + r * 262;
  const float* h = hl + r * 256;
  for (int c = 0; c < 256; ++c) y[c] = h[c];
  const float* hi = hin + r * 13;
  for (int c = 0; c < 5; ++c) y[256 + c] = hi[c];
  y[261] = hi[5];
}

__global__ void k_vprep(const float* hl, const float* hin, const float* xm,
                        const float* nm, const int* na, float* Y)
{
  int b = blockIdx.x, n = threadIdx.x;
  size_t r = (size_t)(b * 64 + n);
  int a = na[b];
  const float* xn = xm + (size_t)(b * 64 + a) * 3;
  const float* xr = xm + r * 3;
  float d0 = xn[0] - xr[0], d1 = xn[1] - xr[1], d2 = xn[2] - xr[2];
  float dist = sqrtf(d0 * d0 + d1 * d1 + d2 * d2);
  float m = nm[b * 64 + n];
  float on = (n == a) ? 1.f : 0.f;
  float* y = Y + r * 265;
  const float* h = hl + r * 256;
  for (int c = 0; c < 256; ++c) y[c] = h[c] * m;
  const float* hi = hin + r * 13;
  for (int c = 0; c < 5; ++c) y[256 + c] = hi[c] * m;
  y[261] = hi[5] * m;
  y[262] = dist * m;
  y[263] = on * m;
  y[264] = (1.f - on) * m;
}

__global__ __launch_bounds__(64) void k_auto(const float* vw, const float* xm,
                                             const float* nm, const int* na,
                                             const float* po, float* out)
{
  __shared__ float sh[3][64];
  int b = blockIdx.x, n = threadIdx.x;
  size_t r = (size_t)(b * 64 + n);
  int a = na[b];
  const float* xn = xm + (size_t)(b * 64 + a) * 3;
  const float* xr = xm + r * 3;
  float m = nm[b * 64 + n];
  float v0 = (xn[0] - xr[0]) * m, v1 = (xn[1] - xr[1]) * m, v2 = (xn[2] - xr[2]) * m;
  float nr = sqrtf(v0 * v0 + v1 * v1 + v2 * v2) + 0.001f;
  float w = vw[r] * __builtin_amdgcn_rcpf(nr);
  sh[0][n] = w * v0; sh[1][n] = w * v1; sh[2][n] = w * v2;
  __syncthreads();
  for (int st = 32; st > 0; st >>= 1) {
    if (n < st) for (int k = 0; k < 3; ++k) sh[k][n] += sh[k][n + st];
    __syncthreads();
  }
  const float* p = po + (size_t)b * 13;
  float am[9], as[9];
  for (int k = 0; k < 3; ++k) { am[k] = xn[k] + sh[k][0]; as[k] = p[0]; }
  for (int q = 0; q < 5; ++q) { am[3 + q] = p[1 + q]; as[3 + q] = p[6 + q]; }
  am[8] = p[11]; as[8] = p[12];
  float* o = out + (size_t)b * OUTROW;
  for (int q = 0; q < 9; ++q) { o[641 + n * 9 + q] = am[q]; o[1217 + n * 9 + q] = as[q]; }
}

// ------------------------------ host helpers -------------------------------
struct Lin { const float* b; const float* w; };
struct GNp { const float* b; const float* g; };
struct AttnBlk { Lin k; GNp n; Lin o, q, v; };
struct ResBlk { Lin conv1, conv2; GNp n1, n2; Lin temb; };
struct EB { Lin att, c1, c2; const float* c3w; Lin edge1, edge2, node1, node2; };
struct PAttn { const unsigned* q; const unsigned* k; const unsigned* v; const unsigned* o; };
struct PRes  { const unsigned* c1; const unsigned* c2; const unsigned* tb; };
struct PEB   { const unsigned* e1a; const unsigned* e1b; const unsigned* e2;
               const unsigned* c1a; const unsigned* c1b; const unsigned* c2;
               const unsigned* n1;  const unsigned* n2; };

static inline void gemm_gen(hipStream_t s, const float* X, const float* W,
                            const float* bias, float* Y, int M, int K, int N,
                            int act = 0, const float* res = nullptr,
                            const float* rm = nullptr)
{
  dim3 g((M + 63) / 64, (N + 63) / 64);
  k_gemm<<<g, 256, 0, s>>>(X, W, bias, res, rm, Y, M, K, N, act);
}
// fast path dispatch: combos used are (bias,act,res,mask) =
// (0,0,0,0) / (1,1,0,0) / (1,0,1,1) / (1,0,1,0) / (1,0,0,0)
static inline void gemm_fast(hipStream_t s, const float* X, const unsigned* Wp,
                             const float* bias, float* Y, int M, int K, int N,
                             int act = 0, const float* res = nullptr,
                             const float* rm = nullptr)
{
  dim3 g(M / 64, N / 64);
  if (!bias)          k_gemm_fast<false,0,false,false><<<g,256,0,s>>>(X,Wp,bias,res,rm,Y,M,K,N);
  else if (act == 1)  k_gemm_fast<true, 1,false,false><<<g,256,0,s>>>(X,Wp,bias,res,rm,Y,M,K,N);
  else if (res && rm) k_gemm_fast<true, 0,true, true ><<<g,256,0,s>>>(X,Wp,bias,res,rm,Y,M,K,N);
  else if (res)       k_gemm_fast<true, 0,true, false><<<g,256,0,s>>>(X,Wp,bias,res,rm,Y,M,K,N);
  else                k_gemm_fast<true, 0,false,false><<<g,256,0,s>>>(X,Wp,bias,res,rm,Y,M,K,N);
}

static void run_trunk(hipStream_t s, const ResBlk* rb, const PRes* rp,
                      const AttnBlk* ab, const PAttn* ap,
                      float** hcur, float* hp, float* gbuf, float* rbuf,
                      float* qb, float* kb, float* vb, float* ob,
                      float* tv, const float* stemb)
{
  float* h = *hcur;
  for (int i = 0; i < 2; ++i) {
    k_gnorm<<<64 * 32, 64, 0, s>>>(h, rb[i].n1.g, rb[i].n1.b, gbuf, 1);
    gemm_fast(s, gbuf, rp[i].c1, rb[i].conv1.b, rbuf, 4096, 256, 256);
    gemm_fast(s, stemb, rp[i].tb, rb[i].temb.b, tv, 64, 256, 256);
    k_addrow<<<4096, 256, 0, s>>>(rbuf, tv, 1048576);
    k_gnorm<<<64 * 32, 64, 0, s>>>(rbuf, rb[i].n2.g, rb[i].n2.b, gbuf, 1);
    gemm_fast(s, gbuf, rp[i].c2, rb[i].conv2.b, hp, 4096, 256, 256, 0, h, nullptr);
    { float* t = h; h = hp; hp = t; }
    k_gnorm<<<64 * 32, 64, 0, s>>>(h, ab[i].n.g, ab[i].n.b, gbuf, 0);
    gemm_fast(s, gbuf, ap[i].q, ab[i].q.b, qb, 4096, 256, 256);
    gemm_fast(s, gbuf, ap[i].k, ab[i].k.b, kb, 4096, 256, 256);
    gemm_fast(s, gbuf, ap[i].v, ab[i].v.b, vb, 4096, 256, 256);
    k_attn<<<64 * 8, 128, 0, s>>>(qb, kb, vb, ob);
    gemm_fast(s, ob, ap[i].o, ab[i].o.b, hp, 4096, 256, 256, 0, h, nullptr);
    { float* t = h; h = hp; hp = t; }
  }
  *hcur = h;
}

// ------------------------------ entry point --------------------------------
extern "C" void kernel_launch(void* const* d_in, const int* in_sizes, int n_in,
                              void* d_out, int out_size, void* d_ws, size_t ws_size,
                              hipStream_t stream)
{
  (void)in_sizes; (void)n_in; (void)out_size; (void)ws_size;
  const float* P       = (const float*)d_in[0];
  const float* x_in    = (const float*)d_in[1];
  const float* oh_in   = (const float*)d_in[2];
  const float* ch_in   = (const float*)d_in[3];
  const float* ctx_in  = (const float*)d_in[4];
  const float* ts_in   = (const float*)d_in[5];
  const int*   dims_in = (const int*)d_in[6];
  const int*   na_in   = (const int*)d_in[7];
  const float* fr_in   = (const float*)d_in[8];
  float* out = (float*)d_out;

  // ---- parameter offsets (jax pytree: dict keys sorted, lists in order) ---
  size_t off = 0;
  auto take = [&](size_t n) { const float* p = P + off; off += n; return p; };
  auto lin  = [&](int din, int dout) { Lin L; L.b = take(dout); L.w = take((size_t)din * dout); return L; };
  auto gn   = [&](int c) { GNp G; G.b = take(c); G.g = take(c); return G; };
  auto attnblk = [&]() { AttnBlk a; a.k = lin(256, 256); a.n = gn(256);
                         a.o = lin(256, 256); a.q = lin(256, 256); a.v = lin(256, 256); return a; };
  auto resblk  = [&]() { ResBlk r; r.conv1 = lin(256, 256); r.conv2 = lin(256, 256);
                         r.n1 = gn(256); r.n2 = gn(256); r.temb = lin(256, 256); return r; };

  AttnBlk attn[2]; attn[0] = attnblk(); attn[1] = attnblk();
  EB blocks[4];
  for (int l = 0; l < 4; ++l) {
    blocks[l].att = lin(256, 1);   blocks[l].c1 = lin(514, 256);
    blocks[l].c2  = lin(256, 256); blocks[l].c3w = take(256);
    blocks[l].edge1 = lin(514, 256); blocks[l].edge2 = lin(256, 256);
    blocks[l].node1 = lin(512, 256); blocks[l].node2 = lin(256, 256);
  }
  Lin emb_in  = lin(13, 256);
  Lin emb_out = lin(256, 7);
  Lin nearL   = lin(256, 1);
  Lin post_auto = lin(256, 13);
  Lin post_rate = lin(256, 1);
  Lin pre_auto  = lin(256, 256);
  Lin pre_rate  = lin(256, 256);
  ResBlk res[2]; res[0] = resblk(); res[1] = resblk();
  Lin t1_in    = lin(262, 256);
  Lin temb_net = lin(256, 256);
  AttnBlk vattn[2]; vattn[0] = attnblk(); vattn[1] = attnblk();
  Lin vec_in = lin(265, 256);
  Lin vec_w  = lin(256, 1);
  ResBlk vres[2]; vres[0] = resblk(); vres[1] = resblk();

  // ------------------------- workspace carving -----------------------------
  float* wsf = (float*)d_ws;
  size_t wo = 0;
  auto alloc = [&](size_t n) { wo = (wo + 3) & ~(size_t)3; float* p = wsf + wo; wo += n; return p; };
  float* nm    = alloc(4096);
  float* hin   = alloc(4096 * 13);
  float* xm    = alloc(12288);
  float* d0m   = alloc(262144);
  float* em    = alloc(262144);
  float* dm    = alloc(262144);
  float* cdiff = alloc(786432);
  float* h     = alloc(1048576);
  float* h2    = alloc(1048576);
  float* Hi    = alloc(1048576);
  float* Hj    = alloc(1048576);
  float* agg   = alloc(1048576);
  float* cat   = alloc(2097152);
  float* tmp   = alloc(1048576);
  float* xagg  = alloc(12288);
  float* xcur  = alloc(12288);
  float* hout  = alloc(28672);
  float* hlast = alloc(1048576);
  float* emb   = alloc(16384);
  float* temb  = alloc(16384);
  float* stemb = alloc(16384);
  float* tv    = alloc(16384);
  float* gbuf  = alloc(1048576);
  float* rbuf  = alloc(1048576);
  float* qb    = alloc(1048576);
  float* kb    = alloc(1048576);
  float* vb    = alloc(1048576);
  float* ob    = alloc(1048576);
  float* t1c   = alloc(4096 * 262);
  float* vcat  = alloc(4096 * 265);
  float* mbuf  = alloc(16384);
  float* pob   = alloc(832);
  float* vwb   = alloc(4096);

  // ---- weight pre-pack (f32 -> bf16 pair-dwords), stream-ordered ----------
  auto packW = [&](const float* w, int K, int N) {
    unsigned* dst = (unsigned*)alloc((size_t)(K / 2) * N);
    int total = (K / 2) * N;
    k_pack<<<(total + 255) / 256, 256, 0, stream>>>(w, dst, K, N);
    return (const unsigned*)dst;
  };
  PEB pblk[4];
  for (int l = 0; l < 4; ++l) {
    pblk[l].e1a = packW(blocks[l].edge1.w, 256, 256);
    pblk[l].e1b = packW(blocks[l].edge1.w + 256 * 256, 256, 256);
    pblk[l].e2  = packW(blocks[l].edge2.w, 256, 256);
    pblk[l].c1a = packW(blocks[l].c1.w, 256, 256);
    pblk[l].c1b = packW(blocks[l].c1.w + 256 * 256, 256, 256);
    pblk[l].c2  = packW(blocks[l].c2.w, 256, 256);
    pblk[l].n1  = packW(blocks[l].node1.w, 512, 256);
    pblk[l].n2  = packW(blocks[l].node2.w, 256, 256);
  }
  PAttn pattn[2], pvattn[2];
  PRes  pres[2],  pvres[2];
  for (int i = 0; i < 2; ++i) {
    pattn[i]  = { packW(attn[i].q.w, 256, 256),  packW(attn[i].k.w, 256, 256),
                  packW(attn[i].v.w, 256, 256),  packW(attn[i].o.w, 256, 256) };
    pvattn[i] = { packW(vattn[i].q.w, 256, 256), packW(vattn[i].k.w, 256, 256),
                  packW(vattn[i].v.w, 256, 256), packW(vattn[i].o.w, 256, 256) };
    pres[i]   = { packW(res[i].conv1.w, 256, 256),  packW(res[i].conv2.w, 256, 256),
                  packW(res[i].temb.w, 256, 256) };
    pvres[i]  = { packW(vres[i].conv1.w, 256, 256), packW(vres[i].conv2.w, 256, 256),
                  packW(vres[i].temb.w, 256, 256) };
  }
  const unsigned* ptemb_net = packW(temb_net.w, 256, 256);
  const unsigned* ppre_rate = packW(pre_rate.w, 256, 256);
  const unsigned* ppre_auto = packW(pre_auto.w, 256, 256);

  // ------------------------------ pipeline ---------------------------------
  k_prep<<<64, 64, 0, stream>>>(x_in, oh_in, ch_in, ctx_in, ts_in, dims_in, nm, hin, xm);
  k_copy<<<48, 256, 0, stream>>>(xm, xcur, 12288);
  k_pair<<<4096, 64, 0, stream>>>(xm, nm, d0m, nullptr, em, 0);
  gemm_gen(stream, hin, emb_in.w, emb_in.b, h, 4096, 13, 256);

  for (int l = 0; l < 4; ++l) {
    const EB& bk = blocks[l];
    const PEB& pk = pblk[l];
    k_pair<<<4096, 64, 0, stream>>>(xcur, nm, dm, cdiff, em, 1);
    // message path (old h)
    gemm_fast(stream, h, pk.e1a, nullptr, Hi, 4096, 256, 256);
    gemm_fast(stream, h, pk.e1b, nullptr, Hj, 4096, 256, 256);
    k_edge<<<4096, 256, 0, stream>>>(Hi, Hj, bk.edge1.w + 512 * 256,
                                     bk.edge1.w + 513 * 256, bk.edge1.b,
                                     pk.e2, bk.edge2.b, bk.att.w, bk.att.b,
                                     dm, d0m, em, nullptr, agg, 0);
    k_cat2<<<(4096 * 512 + 255) / 256, 256, 0, stream>>>(h, agg, cat, 4096);
    gemm_fast(stream, cat, pk.n1, bk.node1.b, tmp, 4096, 512, 256, 1);
    gemm_fast(stream, tmp, pk.n2, bk.node2.b, h2, 4096, 256, 256, 0, h, nm);
    { float* t = h; h = h2; h2 = t; }
    // coordinate path (new h)
    gemm_fast(stream, h, pk.c1a, nullptr, Hi, 4096, 256, 256);
    gemm_fast(stream, h, pk.c1b, nullptr, Hj, 4096, 256, 256);
    k_edge<<<4096, 256, 0, stream>>>(Hi, Hj, bk.c1.w + 512 * 256,
                                     bk.c1.w + 513 * 256, bk.c1.b,
                                     pk.c2, bk.c2.b, bk.c3w, nullptr,
                                     dm, d0m, em, cdiff, xagg, 1);
    k_xupd<<<48, 256, 0, stream>>>(xcur, xagg, nm, 12288);
  }

  k_maskmul<<<4096, 256, 0, stream>>>(h, nm, hlast, 1048576);
  gemm_gen(stream, h, emb_out.w, emb_out.b, hout, 4096, 256, 7, 0, nullptr, nm);
  k_vel<<<64, 64, 0, stream>>>(xcur, xm, nm, dims_in, hout, out);

  // time embedding
  k_tembed<<<64, 128, 0, stream>>>(ts_in, emb);
  gemm_fast(stream, emb, ptemb_net, temb_net.b, temb, 64, 256, 256);
  k_silu<<<64, 256, 0, stream>>>(temb, stemb, 16384);

  // trunk 1 (res/attn)
  k_t1cat<<<64, 64, 0, stream>>>(hlast, hin, t1c);
  gemm_gen(stream, t1c, t1_in.w, t1_in.b, h, 4096, 262, 256);
  run_trunk(stream, res, pres, attn, pattn, &h, h2, gbuf, rbuf, qb, kb, vb, ob, tv, stemb);
  float* pong = (h == h2) ? hlast : h2;   // even # swaps keeps h != h2; pong stays valid
  if (h != h2) pong = h2;

  gemm_fast(stream, h, ppre_rate, pre_rate.b, rbuf, 4096, 256, 256);
  k_mean<<<64, 256, 0, stream>>>(rbuf, mbuf);
  k_rate<<<64, 64, 0, stream>>>(mbuf, post_rate.w, post_rate.b, fr_in, out);
  k_near<<<64, 64, 0, stream>>>(h, nearL.w, nearL.b, out);

  // trunk 2 (vres/vattn)
  k_vprep<<<64, 64, 0, stream>>>(hlast, hin, xm, nm, na_in, vcat);
  gemm_gen(stream, vcat, vec_in.w, vec_in.b, h, 4096, 265, 256);
  run_trunk(stream, vres, pvres, vattn, pvattn, &h, pong, gbuf, rbuf, qb, kb, vb, ob, tv, stemb);

  k_dotrow<<<64, 64, 0, stream>>>(h, vec_w.w, vec_w.b, vwb);
  gemm_fast(stream, h, ppre_auto, pre_auto.b, rbuf, 4096, 256, 256);
  k_mean<<<64, 256, 0, stream>>>(rbuf, mbuf);
  gemm_gen(stream, mbuf, post_auto.w, post_auto.b, pob, 64, 256, 13);
  k_auto<<<64, 64, 0, stream>>>(vwb, xm, nm, na_in, pob, out);
}
